// Multi_head_SelfAttentionPredictModel_2_68607807586677
// MI455X (gfx1250) — matrix operand order repaired and checked by reading, not compiled, not run on hardware
//
#include <hip/hip_runtime.h>
#include <hip/hip_bf16.h>
#include <math.h>

typedef __bf16 bf16;
typedef __attribute__((ext_vector_type(16))) __bf16 v16bf;
typedef __attribute__((ext_vector_type(8)))  __bf16 v8bf;
typedef __attribute__((ext_vector_type(2)))  __bf16 v2bf;
typedef __attribute__((ext_vector_type(8)))  float  v8f;

#define BB   2
#define SS   2048
#define HH   1024
#define NHH  16
#define DHH  64

// ---------------- WMMA helpers (CDNA5 bf16 16x16x32, fp32 accum) ----------------

static __device__ __forceinline__ v16bf cat8(v8bf lo, v8bf hi) {
  return __builtin_shufflevector(lo, hi, 0,1,2,3,4,5,6,7,8,9,10,11,12,13,14,15);
}

// A operand: 16(M) x 32(K) slice of row-major [*, lda] bf16.
static __device__ __forceinline__ v16bf load_a(const bf16* A, int lda, int row0, int k0, int lane) {
  int h = lane >> 4, r = lane & 15;
  const bf16* p = A + (size_t)(row0 + r) * lda + k0 + h * 8;
  return cat8(*(const v8bf*)p, *(const v8bf*)(p + 16));
}

// B operand: 32(K) x 16(N) tile given B^T row-major [N, ldb].
static __device__ __forceinline__ v16bf load_b(const bf16* Bt, int ldb, int n0, int k0, int lane) {
  int h = lane >> 4, c = lane & 15;
  const bf16* p = Bt + (size_t)(n0 + c) * ldb + k0 + h * 8;
  return cat8(*(const v8bf*)p, *(const v8bf*)(p + 16));
}

static __device__ __forceinline__ v8f wmma_bf16(v16bf a, v16bf b, v8f c) {
  return __builtin_amdgcn_wmma_f32_16x16x32_bf16(false, a, false, b, (short)0, c, false, false);
}

// 32(M) x 64(N) tile, software-pipelined K loop (prefetch next K-step before WMMAs).
static __device__ __forceinline__ void gemm_32x64(const bf16* A, int lda, const bf16* Bt, int ldb,
                                                  int K, int m0, int n0, int lane,
                                                  v8f accLo[4], v8f accHi[4]) {
  v16bf a0 = load_a(A, lda, m0,      0, lane);
  v16bf a1 = load_a(A, lda, m0 + 16, 0, lane);
  v16bf b0 = load_b(Bt, ldb, n0 +  0, 0, lane);
  v16bf b1 = load_b(Bt, ldb, n0 + 16, 0, lane);
  v16bf b2 = load_b(Bt, ldb, n0 + 32, 0, lane);
  v16bf b3 = load_b(Bt, ldb, n0 + 48, 0, lane);
  int k0 = 0;
  while (k0 + 32 < K) {
    int kn = k0 + 32;
    // prefetch next K-step (issued before the waits for the current tiles)
    v16bf na0 = load_a(A, lda, m0,      kn, lane);
    v16bf na1 = load_a(A, lda, m0 + 16, kn, lane);
    v16bf nb0 = load_b(Bt, ldb, n0 +  0, kn, lane);
    v16bf nb1 = load_b(Bt, ldb, n0 + 16, kn, lane);
    v16bf nb2 = load_b(Bt, ldb, n0 + 32, kn, lane);
    v16bf nb3 = load_b(Bt, ldb, n0 + 48, kn, lane);
    accLo[0] = wmma_bf16(a0, b0, accLo[0]);
    accHi[0] = wmma_bf16(a1, b0, accHi[0]);
    accLo[1] = wmma_bf16(a0, b1, accLo[1]);
    accHi[1] = wmma_bf16(a1, b1, accHi[1]);
    accLo[2] = wmma_bf16(a0, b2, accLo[2]);
    accHi[2] = wmma_bf16(a1, b2, accHi[2]);
    accLo[3] = wmma_bf16(a0, b3, accLo[3]);
    accHi[3] = wmma_bf16(a1, b3, accHi[3]);
    a0 = na0; a1 = na1; b0 = nb0; b1 = nb1; b2 = nb2; b3 = nb3;
    k0 = kn;
  }
  accLo[0] = wmma_bf16(a0, b0, accLo[0]);
  accHi[0] = wmma_bf16(a1, b0, accHi[0]);
  accLo[1] = wmma_bf16(a0, b1, accLo[1]);
  accHi[1] = wmma_bf16(a1, b1, accHi[1]);
  accLo[2] = wmma_bf16(a0, b2, accLo[2]);
  accHi[2] = wmma_bf16(a1, b2, accHi[2]);
  accLo[3] = wmma_bf16(a0, b3, accLo[3]);
  accHi[3] = wmma_bf16(a1, b3, accHi[3]);
}

// packed 2-row bf16 store for transposed (column-strided) outputs
static __device__ __forceinline__ void store_t_pair(bf16* out, int ld, int n, int m, float lo, float hi) {
  v2bf pk; pk[0] = (bf16)lo; pk[1] = (bf16)hi;
  *(v2bf*)(out + (size_t)n * ld + m) = pk;
}

// ---------------- conversion kernels ----------------

__global__ void k_cvt(const float* __restrict__ src, bf16* __restrict__ dst, int n) {
  for (int i = blockIdx.x * blockDim.x + threadIdx.x; i < n; i += gridDim.x * blockDim.x)
    dst[i] = (bf16)src[i];
}

__global__ void k_cvt_t(const float* __restrict__ src, bf16* __restrict__ dst, int rows, int cols) {
  int z = blockIdx.z;
  const float* s = src + (size_t)z * rows * cols;
  bf16* d = dst + (size_t)z * rows * cols;
  int n = rows * cols;
  for (int i = blockIdx.x * blockDim.x + threadIdx.x; i < n; i += gridDim.x * blockDim.x) {
    int r = i / cols, c = i % cols;
    d[(size_t)c * rows + r] = (bf16)s[i];
  }
}

// ---------------- combined weights: Wc[n] = W_x @ W_xm[n], stored transposed [DH,H] ----------------

__global__ void __launch_bounds__(256)
k_combine(const bf16* __restrict__ Wq, const bf16* __restrict__ Wk, const bf16* __restrict__ Wv,
          const bf16* __restrict__ Wqmt, const bf16* __restrict__ Wkmt, const bf16* __restrict__ Wvmt,
          bf16* __restrict__ Wqct, bf16* __restrict__ Wkct, bf16* __restrict__ Wvct) {
  int lane = threadIdx.x & 31, wid = threadIdx.x >> 5;
  int z = blockIdx.z;                 // 0=q 1=k 2=v
  int head = blockIdx.y;
  int mtile = blockIdx.x * (blockDim.x >> 5) + wid;   // H/32 = 32 tiles
  const bf16* A  = (z == 0) ? Wq : (z == 1) ? Wk : Wv;
  const bf16* Bt = ((z == 0) ? Wqmt : (z == 1) ? Wkmt : Wvmt) + (size_t)head * DHH * HH;
  bf16* out      = ((z == 0) ? Wqct : (z == 1) ? Wkct : Wvct) + (size_t)head * DHH * HH;
  int m0 = mtile * 32;
  v8f accLo[4] = {}, accHi[4] = {};
  gemm_32x64(A, HH, Bt, HH, HH, m0, 0, lane, accLo, accHi);
  int h = lane >> 4, l16 = lane & 15;
#pragma unroll
  for (int t = 0; t < 4; ++t) {
    int n = t * 16 + l16;
#pragma unroll
    for (int j = 0; j < 8; j += 2) {
      store_t_pair(out, HH, n, m0 + j + 8 * h,      accLo[t][j], accLo[t][j + 1]);
      store_t_pair(out, HH, n, m0 + 16 + j + 8 * h, accHi[t][j], accHi[t][j + 1]);
    }
  }
}

// ---------------- projections: Qm/Km row-major [S,DH], Vm transposed [DH,S] ----------------

__global__ void __launch_bounds__(256)
k_project(const bf16* __restrict__ Xb,
          const bf16* __restrict__ Wqct, const bf16* __restrict__ Wkct, const bf16* __restrict__ Wvct,
          bf16* __restrict__ Qm, bf16* __restrict__ Km, bf16* __restrict__ Vmt) {
  int lane = threadIdx.x & 31, wid = threadIdx.x >> 5;
  int z = blockIdx.z;
  int w = blockIdx.x * (blockDim.x >> 5) + wid;       // [0, B*NH*S/32)
  int qt = w % (SS / 32);
  int bn = w / (SS / 32);
  int b = bn / NHH, n = bn % NHH;
  const bf16* A  = Xb + (size_t)b * SS * HH;
  const bf16* Wc = ((z == 0) ? Wqct : (z == 1) ? Wkct : Wvct) + (size_t)n * DHH * HH;
  int m0 = qt * 32;
  v8f accLo[4] = {}, accHi[4] = {};
  gemm_32x64(A, HH, Wc, HH, HH, m0, 0, lane, accLo, accHi);
  int h = lane >> 4, l16 = lane & 15;
  if (z < 2) {
    bf16* out = ((z == 0) ? Qm : Km) + (size_t)bn * SS * DHH;
#pragma unroll
    for (int t = 0; t < 4; ++t)
#pragma unroll
      for (int j = 0; j < 8; ++j) {
        out[(size_t)(m0 + j + 8 * h) * DHH + t * 16 + l16]      = (bf16)accLo[t][j];
        out[(size_t)(m0 + 16 + j + 8 * h) * DHH + t * 16 + l16] = (bf16)accHi[t][j];
      }
  } else {
    bf16* out = Vmt + (size_t)bn * DHH * SS;
#pragma unroll
    for (int t = 0; t < 4; ++t) {
      int d = t * 16 + l16;
#pragma unroll
      for (int j = 0; j < 8; j += 2) {
        store_t_pair(out, SS, d, m0 + j + 8 * h,      accLo[t][j], accLo[t][j + 1]);
        store_t_pair(out, SS, d, m0 + 16 + j + 8 * h, accHi[t][j], accHi[t][j + 1]);
      }
    }
  }
}

// ---------------- flash attention: one wave per 16-query tile, online softmax, pipelined ----------------

__global__ void __launch_bounds__(256)
k_attention(const bf16* __restrict__ Qm, const bf16* __restrict__ Km,
            const bf16* __restrict__ Vmt, bf16* __restrict__ attn) {
  __shared__ __align__(16) bf16 ldsP[8][16][32];      // per-wave P tile (16 q x 32 keys)
  int lane = threadIdx.x & 31, wid = threadIdx.x >> 5;
  int w = blockIdx.x * (blockDim.x >> 5) + wid;       // [0, B*NH*S/16)
  int qt = w % (SS / 16);
  int bn = w / (SS / 16);
  int b = bn / NHH, n = bn % NHH;
  const bf16* Q = Qm  + (size_t)bn * SS * DHH;        // [S,64] row-major
  const bf16* K = Km  + (size_t)bn * SS * DHH;        // [S,64] row-major = B^T for scores
  const bf16* V = Vmt + (size_t)bn * DHH * SS;        // [64,S] = B^T for P@V
  int q0 = qt * 16;
  int h = lane >> 4, l16 = lane & 15;

  v16bf qa0 = load_a(Q, DHH, q0, 0, lane);
  v16bf qa1 = load_a(Q, DHH, q0, 32, lane);

  v8f o[4] = {};
  float mrow[8], lrow[8];
#pragma unroll
  for (int j = 0; j < 8; ++j) { mrow[j] = -1e30f; lrow[j] = 0.f; }

  // prime K-tile pipeline
  v16bf kb0 = load_b(K, DHH, 0,  0, lane);
  v16bf kb1 = load_b(K, DHH, 0, 32, lane);
  v16bf kb2 = load_b(K, DHH, 16, 0, lane);
  v16bf kb3 = load_b(K, DHH, 16, 32, lane);

  for (int kt = 0; kt < SS; kt += 32) {
    // V tiles for the CURRENT block: issued early, consumed after softmax+LDS
    v16bf vb0 = load_b(V, SS,  0, kt, lane);
    v16bf vb1 = load_b(V, SS, 16, kt, lane);
    v16bf vb2 = load_b(V, SS, 32, kt, lane);
    v16bf vb3 = load_b(V, SS, 48, kt, lane);
    // K tiles for the NEXT block: prefetch
    int ktn = kt + 32;
    v16bf nk0 = kb0, nk1 = kb1, nk2 = kb2, nk3 = kb3;
    if (ktn < SS) {
      nk0 = load_b(K, DHH, ktn,       0, lane);
      nk1 = load_b(K, DHH, ktn,      32, lane);
      nk2 = load_b(K, DHH, ktn + 16,  0, lane);
      nk3 = load_b(K, DHH, ktn + 16, 32, lane);
    }

    // scores: 16 queries x 32 keys, K-dim = DH = 64 (operands already resident)
    v8f s0 = {}, s1 = {};
    s0 = wmma_bf16(qa0, kb0, s0);
    s0 = wmma_bf16(qa1, kb1, s0);
    s1 = wmma_bf16(qa0, kb2, s1);
    s1 = wmma_bf16(qa1, kb3, s1);
    kb0 = nk0; kb1 = nk1; kb2 = nk2; kb3 = nk3;

    float p0[8], p1[8];
#pragma unroll
    for (int j = 0; j < 8; ++j) {
      float t = fmaxf(s0[j], s1[j]);
      t = fmaxf(t, __shfl_xor(t, 1, 32));
      t = fmaxf(t, __shfl_xor(t, 2, 32));
      t = fmaxf(t, __shfl_xor(t, 4, 32));
      t = fmaxf(t, __shfl_xor(t, 8, 32));
      float mn = fmaxf(mrow[j], t);
      float scale = __expf(mrow[j] - mn);
      mrow[j] = mn;
      p0[j] = __expf(s0[j] - mn);
      p1[j] = __expf(s1[j] - mn);
      float r = p0[j] + p1[j];
      r += __shfl_xor(r, 1, 32);
      r += __shfl_xor(r, 2, 32);
      r += __shfl_xor(r, 4, 32);
      r += __shfl_xor(r, 8, 32);
      lrow[j] = lrow[j] * scale + r;
#pragma unroll
      for (int t4 = 0; t4 < 4; ++t4) o[t4][j] *= scale;
    }

    // C/D layout -> A layout via LDS (row-major 16x32 bf16)
#pragma unroll
    for (int j = 0; j < 8; ++j) {
      ldsP[wid][j + 8 * h][l16]      = (bf16)p0[j];
      ldsP[wid][j + 8 * h][16 + l16] = (bf16)p1[j];
    }
    asm volatile("s_wait_dscnt 0" ::: "memory");
    v16bf pa = load_a(&ldsP[wid][0][0], 32, 0, 0, lane);

    o[0] = wmma_bf16(pa, vb0, o[0]);
    o[1] = wmma_bf16(pa, vb1, o[1]);
    o[2] = wmma_bf16(pa, vb2, o[2]);
    o[3] = wmma_bf16(pa, vb3, o[3]);
  }

  // normalize and store concat-head layout: attn[b][q][n*64 + d]
  bf16* out = attn + ((size_t)b * SS + q0) * HH + n * DHH;
#pragma unroll
  for (int t4 = 0; t4 < 4; ++t4)
#pragma unroll
    for (int j = 0; j < 8; ++j) {
      float v = o[t4][j] / lrow[j];
      out[(size_t)(j + 8 * h) * HH + t4 * 16 + l16] = (bf16)v;
    }
}

// ---------------- output projection: [B*S,H] @ W_o -> fp32 ----------------

__global__ void __launch_bounds__(256)
k_out_gemm(const bf16* __restrict__ attn, const bf16* __restrict__ Wot, float* __restrict__ out) {
  int lane = threadIdx.x & 31, wid = threadIdx.x >> 5;
  int w = blockIdx.x * (blockDim.x >> 5) + wid;   // (B*S/32) * (H/64) = 2048 waves
  int nt = w % (HH / 64);
  int mt = w / (HH / 64);
  int m0 = mt * 32, n0 = nt * 64;
  v8f accLo[4] = {}, accHi[4] = {};
  gemm_32x64(attn, HH, Wot, HH, HH, m0, n0, lane, accLo, accHi);
  int h = lane >> 4, l16 = lane & 15;
#pragma unroll
  for (int t = 0; t < 4; ++t)
#pragma unroll
    for (int j = 0; j < 8; ++j) {
      out[(size_t)(m0 + j + 8 * h) * HH + n0 + t * 16 + l16]      = accLo[t][j];
      out[(size_t)(m0 + 16 + j + 8 * h) * HH + n0 + t * 16 + l16] = accHi[t][j];
    }
}

// ---------------- host ----------------

extern "C" void kernel_launch(void* const* d_in, const int* in_sizes, int n_in,
                              void* d_out, int out_size, void* d_ws, size_t ws_size,
                              hipStream_t stream) {
  const float* X    = (const float*)d_in[0];
  const float* W_q  = (const float*)d_in[1];
  const float* W_k  = (const float*)d_in[2];
  const float* W_v  = (const float*)d_in[3];
  const float* W_qm = (const float*)d_in[4];
  const float* W_km = (const float*)d_in[5];
  const float* W_vm = (const float*)d_in[6];
  const float* W_o  = (const float*)d_in[7];
  float* out = (float*)d_out;

  bf16* p = (bf16*)d_ws;
  bf16* Xb    = p; p += (size_t)BB * SS * HH;
  bf16* Wqb   = p; p += (size_t)HH * HH;
  bf16* Wkb   = p; p += (size_t)HH * HH;
  bf16* Wvb   = p; p += (size_t)HH * HH;
  bf16* Wqmt  = p; p += (size_t)NHH * DHH * HH;
  bf16* Wkmt  = p; p += (size_t)NHH * DHH * HH;
  bf16* Wvmt  = p; p += (size_t)NHH * DHH * HH;
  bf16* Wot   = p; p += (size_t)HH * HH;
  bf16* Wqct  = p; p += (size_t)NHH * DHH * HH;
  bf16* Wkct  = p; p += (size_t)NHH * DHH * HH;
  bf16* Wvct  = p; p += (size_t)NHH * DHH * HH;
  bf16* Qm    = p; p += (size_t)BB * NHH * SS * DHH;
  bf16* Km    = p; p += (size_t)BB * NHH * SS * DHH;
  bf16* Vmt   = p; p += (size_t)BB * NHH * DHH * SS;
  bf16* attnb = p; p += (size_t)BB * SS * HH;

  k_cvt<<<2048, 256, 0, stream>>>(X,   Xb,  BB * SS * HH);
  k_cvt<<<1024, 256, 0, stream>>>(W_q, Wqb, HH * HH);
  k_cvt<<<1024, 256, 0, stream>>>(W_k, Wkb, HH * HH);
  k_cvt<<<1024, 256, 0, stream>>>(W_v, Wvb, HH * HH);
  k_cvt_t<<<dim3(256, 1, NHH), 256, 0, stream>>>(W_qm, Wqmt, HH, DHH);
  k_cvt_t<<<dim3(256, 1, NHH), 256, 0, stream>>>(W_km, Wkmt, HH, DHH);
  k_cvt_t<<<dim3(256, 1, NHH), 256, 0, stream>>>(W_vm, Wvmt, HH, DHH);
  k_cvt_t<<<dim3(1024, 1, 1),  256, 0, stream>>>(W_o,  Wot,  HH, HH);

  k_combine<<<dim3(4, NHH, 3), 256, 0, stream>>>(Wqb, Wkb, Wvb, Wqmt, Wkmt, Wvmt,
                                                 Wqct, Wkct, Wvct);

  k_project<<<dim3(256, 1, 3), 256, 0, stream>>>(Xb, Wqct, Wkct, Wvct, Qm, Km, Vmt);

  k_attention<<<dim3(512), 256, 0, stream>>>(Qm, Km, Vmt, attnb);

  k_out_gemm<<<dim3(256), 256, 0, stream>>>(attnb, Wot, out);
}
